// VQVAE_56075093017129
// MI455X (gfx1250) — matrix operand order repaired and checked by reading, not compiled, not run on hardware
//
#include <hip/hip_runtime.h>
#include <hip/hip_bf16.h>

typedef __attribute__((ext_vector_type(16))) _Float16 v16h;
typedef __attribute__((ext_vector_type(8)))  float    v8f;

#define EMB_K 8
#define BATCH 64

// ---------------------------------------------------------------------------
// small utility kernels
// ---------------------------------------------------------------------------
__global__ void zero_losses_k(float* out) {
  if (threadIdx.x < 3) out[threadIdx.x] = 0.0f;
}

__global__ void cvt_f32_f16_k(const float* __restrict__ s,
                              _Float16* __restrict__ d, int n) {
  int i  = blockIdx.x * blockDim.x + threadIdx.x;
  int st = gridDim.x * blockDim.x;
  for (; i < n; i += st) d[i] = (_Float16)s[i];
}

// ---------------------------------------------------------------------------
// Pack conv weights into WMMA A-fragment layout (f16):
//   dst[((kt*Mtiles + mt)*32 + lane)*16 + h]
// A 16x32 f16 lane layout (ISA 7.12.2):
//   lanes 0-15 : M = lane,    K = h (h<8) / h+8  (h>=8)
//   lanes 16-31: M = lane-16, K = h+8 (h<8) / h+16 (h>=8)
// flip==1: ConvTranspose weights (I,O,kh,kw) -> flipped OIHW
// (one-shot tiny kernel; runtime divides are irrelevant here)
// ---------------------------------------------------------------------------
__global__ void pack_w_k(const float* __restrict__ w, _Float16* __restrict__ dst,
                         int Cout, int Cin, int kh, int kw,
                         int Mtiles, int Ksteps, int flip) {
  int idx   = blockIdx.x * blockDim.x + threadIdx.x;
  int total = Ksteps * Mtiles * 512;
  if (idx >= total) return;
  int kt = idx / (Mtiles * 512);
  int r  = idx - kt * (Mtiles * 512);
  int mt = r >> 9;
  int r2 = r & 511;
  int l  = r2 >> 4;   // lane 0..31
  int h  = r2 & 15;   // half 0..15
  int m  = mt * 16 + (l & 15);
  int kloc = (l < 16) ? ((h < 8) ? h : h + 8)
                      : ((h < 8) ? h + 8 : h + 16);
  int k   = kt * 32 + kloc;
  int khw = kh * kw;
  float v = 0.0f;
  if (k < Cin * khw && m < Cout) {
    int ci = k / khw;
    int rr = k - ci * khw;
    int ky = rr / kw;
    int kx = rr - ky * kw;
    if (flip)
      v = w[(((size_t)ci * Cout + m) * kh + (kh - 1 - ky)) * kw + (kw - 1 - kx)];
    else
      v = w[(((size_t)m * Cin + ci) * kh + ky) * kw + kx];
  }
  dst[idx] = (_Float16)v;
}

// ---------------------------------------------------------------------------
// Fully-templated implicit-GEMM convolution via v_wmma_f32_16x16x32_f16.
//   M = COUT (one 16-row tile per wave, MT waves per block)
//   N = NT*16 output pixels per block
//   K = CIN*KH*KW padded up to multiple of 32
// All geometry compile-time -> address math is shifts / magic multiplies.
// Double-buffered fragment-packed B tiles in LDS: one barrier per K-step,
// next tile's global gathers issued before the current tile's WMMAs.
// Transposed conv: STRIDE=1, PAD=k-1, DIL=lhs_dilation, weights pre-flipped.
// ---------------------------------------------------------------------------
template <int CIN, int IH, int IW, int COUT, int OH, int OW,
          int KH, int KW, int STRIDE, int PAD, int DIL, int RELU, int NT>
__global__ void __launch_bounds__(32 * ((COUT + 15) / 16))
conv_wmma_k(const _Float16* __restrict__ in,
            const _Float16* __restrict__ wPack,
            const float*    __restrict__ bias,
            _Float16* __restrict__ out16,
            float*    __restrict__ out32, int write32) {
  constexpr int KHW    = KH * KW;
  constexpr int KTOT   = CIN * KHW;
  constexpr int KSTEPS = (KTOT + 31) / 32;
  constexpr int MT     = (COUT + 15) / 16;
  constexpr int OHW    = OH * OW;
  constexpr int NPIX   = BATCH * OHW;
  constexpr int ELEMS  = NT * 512;          // halves per LDS buffer

  __shared__ v16h lb[2][NT * 32];           // 2 x NT*16-col fragment-packed B

  const int tid   = threadIdx.x;
  const int lane  = tid & 31;
  const int mt    = tid >> 5;               // wave id == M tile
  const int nbase = blockIdx.x * (NT * 16);

  // fragment-packed im2col gather of one 32-deep K slice into LDS buffer
  auto fill = [&](int buf, int kt) {
    for (int idx = tid; idx < ELEMS; idx += 32 * MT) {
      int ntile = idx >> 9;
      int r     = idx & 511;
      int l     = r >> 4;
      int h     = r & 15;
      int nloc  = l & 15;
      int kloc  = ((l >> 4) << 3) + ((h < 8) ? h : h + 8);  // +8 for lanes 16..31
      int n     = nbase + ntile * 16 + nloc;
      int k     = kt * 32 + kloc;
      _Float16 v = (_Float16)0.0f;
      if (n < NPIX && k < KTOT) {
        int ci = k / KHW;
        int rr = k - ci * KHW;
        int ky = rr / KW;
        int kx = rr - ky * KW;
        int b  = n / OHW;
        int p  = n - b * OHW;
        int oy = p / OW;
        int ox = p - oy * OW;
        int iy = oy * STRIDE + ky - PAD;
        int ix = ox * STRIDE + kx - PAD;
        bool ok = true;
        if (DIL != 1) {
          ok = (iy >= 0) && (ix >= 0) && (iy % DIL == 0) && (ix % DIL == 0);
          if (ok) { iy /= DIL; ix /= DIL; }
        }
        if (ok && (unsigned)iy < (unsigned)IH && (unsigned)ix < (unsigned)IW)
          v = in[(((size_t)b * CIN + ci) * IH + iy) * IW + ix];
      }
      ((_Float16*)lb[buf])[idx] = v;
    }
  };

  const v8f zf = {0.f, 0.f, 0.f, 0.f, 0.f, 0.f, 0.f, 0.f};
  v8f acc[NT];
#pragma unroll
  for (int nt = 0; nt < NT; ++nt) acc[nt] = zf;

  fill(0, 0);
  for (int kt = 0; kt < KSTEPS; ++kt) {
    __syncthreads();                        // buffer kt&1 ready; prev reads done
    if (kt + 1 < KSTEPS) {
      fill((kt + 1) & 1, kt + 1);           // overlap next gather with WMMAs
      __builtin_prefetch(
          wPack + (size_t)((kt + 1) * MT + mt) * 32 * 16, 0, 0);
    }
    // A fragment: contiguous pre-packed 32B load per lane (L2-resident)
    const v16h af =
        *(const v16h*)(wPack + ((size_t)(kt * MT + mt) * 32 + lane) * 16);
#pragma unroll
    for (int nt = 0; nt < NT; ++nt) {
      const v16h bf = lb[kt & 1][nt * 32 + lane];
      acc[nt] = __builtin_amdgcn_wmma_f32_16x16x32_f16(
          /*neg_a=*/false, af, /*neg_b=*/false, bf,
          /*c_mod=*/(short)0, acc[nt],
          /*reuse_a=*/false, /*reuse_b=*/false);
    }
  }

  // epilogue: bias + relu; C/D layout: VGPR r -> M = r (+8 for lanes>=16)
#pragma unroll
  for (int nt = 0; nt < NT; ++nt) {
    int n = nbase + nt * 16 + (lane & 15);
    if (n >= NPIX) continue;
    int b = n / OHW;
    int p = n - b * OHW;
#pragma unroll
    for (int r = 0; r < 8; ++r) {
      int ch = mt * 16 + r + ((lane >> 4) << 3);
      if (ch < COUT) {
        float v = acc[nt][r] + bias[ch];
        if (RELU) v = fmaxf(v, 0.0f);
        size_t o = ((size_t)b * COUT + ch) * OHW + p;
        out16[o] = (_Float16)v;
        if (write32) out32[o] = v;
      }
    }
  }
}

// ---------------------------------------------------------------------------
// Vector quantization: argmax of (||z||^2 - 2 z.e + ||e||^2) over 8 codes
// (faithful to the reference's argmax, first-max tie break). Writes index
// (as float) into out[3+n], codebook vectors (f16) for the decoder, and
// accumulates dict/enc losses into out[1], out[2].
// ---------------------------------------------------------------------------
__global__ void vq_k(const float* __restrict__ z, const float* __restrict__ emb,
                     _Float16* __restrict__ val16, float* __restrict__ out,
                     float lscale) {
  constexpr int C  = 128;
  constexpr int HW = 27 * 27;
  constexpr int NP = BATCH * HW;

  __shared__ float se[EMB_K * C];
  __shared__ float see[EMB_K];
  for (int i = threadIdx.x; i < EMB_K * C; i += blockDim.x) se[i] = emb[i];
  __syncthreads();
  if (threadIdx.x < EMB_K) {
    float s = 0.f;
    for (int c = 0; c < C; ++c) {
      float e = se[threadIdx.x * C + c];
      s += e * e;
    }
    see[threadIdx.x] = s;
  }
  __syncthreads();

  int n = blockIdx.x * blockDim.x + threadIdx.x;
  if (n >= NP) return;
  int b = n / HW;
  int p = n - b * HW;
  const float* zb = z + (size_t)b * C * HW + p;

  float dot[EMB_K];
#pragma unroll
  for (int k = 0; k < EMB_K; ++k) dot[k] = 0.f;
  float zz = 0.f;
  for (int c = 0; c < C; ++c) {
    float zc = zb[(size_t)c * HW];
    zz += zc * zc;
#pragma unroll
    for (int k = 0; k < EMB_K; ++k) dot[k] += zc * se[k * C + c];
  }
  int best = 0;
  float bv = -3.4e38f;
#pragma unroll
  for (int k = 0; k < EMB_K; ++k) {
    float d2 = zz - 2.0f * dot[k] + see[k];
    if (d2 > bv) { bv = d2; best = k; }
  }
  out[3 + n] = (float)best;

  float s = 0.f;
  for (int c = 0; c < C; ++c) {
    float e  = se[best * C + c];
    float zc = zb[(size_t)c * HW];
    val16[(size_t)b * C * HW + (size_t)c * HW + p] = (_Float16)e;
    float d = e - zc;
    s += d * d;
  }
  atomicAdd(&out[1], s * lscale);   // dict loss
  atomicAdd(&out[2], s * lscale);   // enc loss (numerically identical)
}

__global__ void rec_loss_k(const float* __restrict__ mu,
                           const float* __restrict__ x,
                           float* __restrict__ out, int n, float scale) {
  int i  = blockIdx.x * blockDim.x + threadIdx.x;
  int st = gridDim.x * blockDim.x;
  float s = 0.f;
  for (; i < n; i += st) {
    float d = mu[i] - x[i];
    s += d * d;
  }
  atomicAdd(&out[0], s * scale);
}

// ---------------------------------------------------------------------------
// Host orchestration
// ---------------------------------------------------------------------------
static inline size_t alignup256(size_t v) { return (v + 255) & ~(size_t)255; }
static inline int cdiv(int a, int b) { return (a + b - 1) / b; }

extern "C" void kernel_launch(void* const* d_in, const int* in_sizes, int n_in,
                              void* d_out, int out_size, void* d_ws, size_t ws_size,
                              hipStream_t stream) {
  (void)in_sizes; (void)n_in; (void)out_size; (void)ws_size;

  const float* x   = (const float*)d_in[0];
  const float* ew0 = (const float*)d_in[1];  const float* eb0 = (const float*)d_in[2];
  const float* ew1 = (const float*)d_in[3];  const float* eb1 = (const float*)d_in[4];
  const float* ew2 = (const float*)d_in[5];  const float* eb2 = (const float*)d_in[6];
  const float* ew3 = (const float*)d_in[7];  const float* eb3 = (const float*)d_in[8];
  const float* dw0 = (const float*)d_in[9];  const float* db0 = (const float*)d_in[10];
  const float* dw1 = (const float*)d_in[11]; const float* db1 = (const float*)d_in[12];
  const float* dw2 = (const float*)d_in[13]; const float* db2 = (const float*)d_in[14];
  const float* dw3 = (const float*)d_in[15]; const float* db3 = (const float*)d_in[16];
  const float* mw0 = (const float*)d_in[17]; const float* mb0 = (const float*)d_in[18];
  const float* mw1 = (const float*)d_in[19]; const float* mb1 = (const float*)d_in[20];
  const float* emb = (const float*)d_in[21];

  float* out = (float*)d_out;
  char*  ws  = (char*)d_ws;

  const int B = BATCH;
  size_t off = 0;
  auto alloc = [&](size_t bytes) -> char* {
    char* p = ws + off;
    off = alignup256(off + bytes);
    return p;
  };

  _Float16* x16   = (_Float16*)alloc((size_t)B * 64 * 64 * 2);
  _Float16* ping  = (_Float16*)alloc((size_t)B * 128 * 62 * 62 * 2);
  _Float16* pong  = (_Float16*)alloc((size_t)B * 128 * 64 * 64 * 2);
  _Float16* z16   = (_Float16*)alloc((size_t)B * 128 * 729 * 2);
  float*    z32   = (float*)   alloc((size_t)B * 128 * 729 * 4);
  _Float16* val16 = (_Float16*)alloc((size_t)B * 128 * 729 * 2);
  float*    mu32  = (float*)   alloc((size_t)B * 64 * 64 * 4);
  _Float16* mu16  = (_Float16*)alloc((size_t)B * 64 * 64 * 2);

  auto wsz = [](int Cin, int kh, int kw, int Cout) -> size_t {
    int Ks = (Cin * kh * kw + 31) / 32;
    int Mt = (Cout + 15) / 16;
    return (size_t)Ks * Mt * 512 * 2;
  };
  _Float16* wp0 = (_Float16*)alloc(wsz(1,   3, 3, 128));
  _Float16* wp1 = (_Float16*)alloc(wsz(128, 3, 3, 128));
  _Float16* wp2 = (_Float16*)alloc(wsz(128, 3, 3, 128));
  _Float16* wp3 = (_Float16*)alloc(wsz(128, 3, 3, 128));
  _Float16* wp4 = (_Float16*)alloc(wsz(128, 3, 3, 128));
  _Float16* wp5 = (_Float16*)alloc(wsz(128, 3, 3, 128));
  _Float16* wp6 = (_Float16*)alloc(wsz(128, 4, 4, 128));
  _Float16* wp7 = (_Float16*)alloc(wsz(128, 3, 3, 128));
  _Float16* wp8 = (_Float16*)alloc(wsz(128, 1, 1, 64));
  _Float16* wp9 = (_Float16*)alloc(wsz(64,  1, 1, 1));

  zero_losses_k<<<1, 32, 0, stream>>>(out);
  cvt_f32_f16_k<<<256, 256, 0, stream>>>(x, x16, B * 64 * 64);

  auto pack = [&](const float* w, _Float16* dst, int Cout, int Cin,
                  int kh, int kw, int flip) {
    int Ks = (Cin * kh * kw + 31) / 32;
    int Mt = (Cout + 15) / 16;
    int total = Ks * Mt * 512;
    pack_w_k<<<cdiv(total, 256), 256, 0, stream>>>(w, dst, Cout, Cin, kh, kw,
                                                   Mt, Ks, flip);
  };
  pack(ew0, wp0, 128, 1,   3, 3, 0);
  pack(ew1, wp1, 128, 128, 3, 3, 0);
  pack(ew2, wp2, 128, 128, 3, 3, 0);
  pack(ew3, wp3, 128, 128, 3, 3, 0);
  pack(dw0, wp4, 128, 128, 3, 3, 1);
  pack(dw1, wp5, 128, 128, 3, 3, 1);
  pack(dw2, wp6, 128, 128, 4, 4, 1);
  pack(dw3, wp7, 128, 128, 3, 3, 1);
  pack(mw0, wp8, 64,  128, 1, 1, 0);
  pack(mw1, wp9, 1,   64,  1, 1, 0);

  // 128 output pixels per block (NT=8); blockDim = 32 * ceil(COUT/16)
  // encoder
  conv_wmma_k<1,   64, 64, 128, 62, 62, 3, 3, 1, 0, 1, 1, 8>
      <<<cdiv(B * 62 * 62, 128), 256, 0, stream>>>(x16, wp0, eb0, ping, nullptr, 0);
  conv_wmma_k<128, 62, 62, 128, 60, 60, 3, 3, 1, 0, 1, 1, 8>
      <<<cdiv(B * 60 * 60, 128), 256, 0, stream>>>(ping, wp1, eb1, pong, nullptr, 0);
  conv_wmma_k<128, 60, 60, 128, 29, 29, 3, 3, 2, 0, 1, 1, 8>
      <<<cdiv(B * 29 * 29, 128), 256, 0, stream>>>(pong, wp2, eb2, ping, nullptr, 0);
  conv_wmma_k<128, 29, 29, 128, 27, 27, 3, 3, 1, 0, 1, 0, 8>
      <<<cdiv(B * 27 * 27, 128), 256, 0, stream>>>(ping, wp3, eb3, z16, z32, 1);

  // vector quantization (+ dict/enc losses, index output)
  {
    int Np = B * 27 * 27;
    float lscale = 0.01f / (float)((size_t)B * 128 * 27 * 27);
    vq_k<<<cdiv(Np, 256), 256, 0, stream>>>(z32, emb, val16, out, lscale);
  }

  // decoder (ConvTranspose as conv: stride=1, pad=k-1, lhs_dilation=DIL)
  conv_wmma_k<128, 27, 27, 128, 29, 29, 3, 3, 1, 2, 1, 1, 8>
      <<<cdiv(B * 29 * 29, 128), 256, 0, stream>>>(val16, wp4, db0, ping, nullptr, 0);
  conv_wmma_k<128, 29, 29, 128, 59, 59, 3, 3, 1, 2, 2, 1, 8>
      <<<cdiv(B * 59 * 59, 128), 256, 0, stream>>>(ping, wp5, db1, pong, nullptr, 0);
  conv_wmma_k<128, 59, 59, 128, 62, 62, 4, 4, 1, 3, 1, 1, 8>
      <<<cdiv(B * 62 * 62, 128), 256, 0, stream>>>(pong, wp6, db2, ping, nullptr, 0);
  conv_wmma_k<128, 62, 62, 128, 64, 64, 3, 3, 1, 2, 1, 1, 8>
      <<<cdiv(B * 64 * 64, 128), 256, 0, stream>>>(ping, wp7, db3, pong, nullptr, 0);

  // mu_net (1x1 convs)
  conv_wmma_k<128, 64, 64, 64, 64, 64, 1, 1, 1, 0, 1, 1, 8>
      <<<cdiv(B * 64 * 64, 128), 128, 0, stream>>>(pong, wp8, mb0, ping, nullptr, 0);
  conv_wmma_k<64,  64, 64, 1,  64, 64, 1, 1, 1, 0, 1, 0, 8>
      <<<cdiv(B * 64 * 64, 128), 32, 0, stream>>>(ping, wp9, mb1, mu16, mu32, 1);

  // reconstruction loss
  {
    int n = B * 64 * 64;
    rec_loss_k<<<256, 256, 0, stream>>>(mu32, x, out, n, 1.0f / (float)n);
  }
}
